// KNNAttention3D_824633721332
// MI455X (gfx1250) — matrix operand order repaired and checked by reading, not compiled, not run on hardware
//
#include <hip/hip_runtime.h>
#include <hip/hip_bf16.h>

// ---------------------------------------------------------------------------
// KNN Attention 3D:  B=1, D=H=W=16 (N=4096), C=528, NH=8, HD=66, K=32
// Pipeline:
//   prep   : fp32 -> bf16 conversions with K padded 528->544, zero attn buf
//   gemm1  : qkv[4096x1584]f32 = Xb @ Wqkv^T + b_qkv   (v_wmma bf16, 32x48 tile)
//   attn   : per-(h,n) wave: l2norm q/k, 33-way online softmax over knn_kv,
//            streaming NT loads; writes bf16 attn-out [4096x544]
//   gemm2  : out[4096x528]f32 = Ab @ Wproj^T + b_proj  (v_wmma bf16, 32x48 tile)
// ---------------------------------------------------------------------------

#define NPOS 4096
#define CIN  528
#define KPAD 544          // 528 padded to multiple of 32
#define CQKV 1584
#define NHEAD 8
#define HD   66
#define KNN  32

typedef __attribute__((ext_vector_type(16))) __bf16 v16bf;
typedef __attribute__((ext_vector_type(8)))  __bf16 v8bf;
typedef __attribute__((ext_vector_type(8)))  float  v8f;

// ---- fragment loaders (row-major, K contiguous, ld in elements) -----------

// A-matrix 16x32 bf16 fragment (ISA 7.12.2): lanes 0-15 -> M=lane,
// v0..3 hold K = 8*hi + {0..7}, v4..7 hold K = 16 + 8*hi + {0..7}
__device__ __forceinline__ v16bf load_fragA(const __hip_bfloat16* __restrict__ base,
                                            int ld, int row0, int k0, int lane) {
  int r  = lane & 15;
  int hi = lane >> 4;
  const __hip_bfloat16* p = base + (size_t)(row0 + r) * ld + k0 + hi * 8;
  v8bf lo  = *(const v8bf*)(p);       // 16B: K = 8*hi + 0..7
  v8bf hi8 = *(const v8bf*)(p + 16);  // 16B: K = 16 + 8*hi + 0..7
  return __builtin_shufflevector(lo, hi8, 0,1,2,3,4,5,6,7,8,9,10,11,12,13,14,15);
}

// B-matrix 32x16 bf16 fragment: lane holds column (lane&15),
// v0..3 hold K = 8*hi + {0..7}, v4..7 hold K = 16 + 8*hi + {0..7}
__device__ __forceinline__ v16bf load_fragB(const __hip_bfloat16* __restrict__ base,
                                            int ld, int n0, int k0, int lane) {
  int c  = lane & 15;
  int hi = lane >> 4;
  const __hip_bfloat16* p = base + (size_t)(n0 + c) * ld + k0 + hi * 8;
  v8bf lo  = *(const v8bf*)(p);       // K = 8*hi + 0..7
  v8bf hi8 = *(const v8bf*)(p + 16);  // K = 16 + 8*hi + 0..7
  return __builtin_shufflevector(lo, hi8, 0,1,2,3,4,5,6,7,8,9,10,11,12,13,14,15);
}

// ---- prep kernels ----------------------------------------------------------

__global__ void cvt_pad_bf16_kernel(const float* __restrict__ src,
                                    __hip_bfloat16* __restrict__ dst,
                                    int rows, int cols, int ldd) {
  int i = blockIdx.x * blockDim.x + threadIdx.x;
  int total = rows * ldd;
  if (i >= total) return;
  int r = i / ldd, c = i - r * ldd;
  float v = (c < cols) ? src[(size_t)r * cols + c] : 0.0f;
  dst[i] = __float2bfloat16(v);
}

__global__ void zero_bf16_kernel(__hip_bfloat16* __restrict__ dst, int total) {
  int i = blockIdx.x * blockDim.x + threadIdx.x;
  if (i < total) dst[i] = __float2bfloat16(0.0f);
}

// ---- WMMA GEMM: out[M x N] = A[M x KPAD] @ W[N x KPAD]^T + bias -----------
// One wave computes a 32x48 tile: 2 A-frags x 3 B-frags = 6 wmma per K-step.

__global__ void gemm_bf16_nt_kernel(const __hip_bfloat16* __restrict__ A,
                                    const __hip_bfloat16* __restrict__ W,
                                    const float* __restrict__ bias,
                                    float* __restrict__ out,
                                    int M, int N, int nStrips) {
  int wave = (int)((blockIdx.x * blockDim.x + threadIdx.x) >> 5);
  int lane = threadIdx.x & 31;
  int totalWaves = (M >> 5) * nStrips;
  if (wave >= totalWaves) return;     // whole-wave uniform: EXEC stays all-ones

  int mtile  = wave / nStrips;
  int nstrip = wave - mtile * nStrips;
  int m0 = mtile << 5;
  int n0 = nstrip * 48;

  v8f c00 = {}; v8f c01 = {}; v8f c02 = {};
  v8f c10 = {}; v8f c11 = {}; v8f c12 = {};
#pragma unroll 1
  for (int k0 = 0; k0 < KPAD; k0 += 32) {
    v16bf a0 = load_fragA(A, KPAD, m0,      k0, lane);
    v16bf a1 = load_fragA(A, KPAD, m0 + 16, k0, lane);
    v16bf b0 = load_fragB(W, KPAD, n0,      k0, lane);
    v16bf b1 = load_fragB(W, KPAD, n0 + 16, k0, lane);
    v16bf b2 = load_fragB(W, KPAD, n0 + 32, k0, lane);
    c00 = __builtin_amdgcn_wmma_f32_16x16x32_bf16(false, a0, false, b0, (short)0, c00, false, false);
    c01 = __builtin_amdgcn_wmma_f32_16x16x32_bf16(false, a0, false, b1, (short)0, c01, false, false);
    c02 = __builtin_amdgcn_wmma_f32_16x16x32_bf16(false, a0, false, b2, (short)0, c02, false, false);
    c10 = __builtin_amdgcn_wmma_f32_16x16x32_bf16(false, a1, false, b0, (short)0, c10, false, false);
    c11 = __builtin_amdgcn_wmma_f32_16x16x32_bf16(false, a1, false, b1, (short)0, c11, false, false);
    c12 = __builtin_amdgcn_wmma_f32_16x16x32_bf16(false, a1, false, b2, (short)0, c12, false, false);
  }

  // C/D layout: lanes 0-15 -> N=lane, M = j; lanes 16-31 -> N=lane-16, M = 8+j
  int col   = lane & 15;
  int rbase = m0 + ((lane >> 4) << 3);
  float bi0 = bias[n0 + col];
  float bi1 = bias[n0 + 16 + col];
  float bi2 = bias[n0 + 32 + col];
#pragma unroll
  for (int j = 0; j < 8; ++j) {
    size_t r0 = (size_t)(rbase + j);
    size_t r1 = r0 + 16;
    out[r0 * N + n0 + col]      = c00[j] + bi0;
    out[r0 * N + n0 + 16 + col] = c01[j] + bi1;
    out[r0 * N + n0 + 32 + col] = c02[j] + bi2;
    out[r1 * N + n0 + col]      = c10[j] + bi0;
    out[r1 * N + n0 + 16 + col] = c11[j] + bi1;
    out[r1 * N + n0 + 32 + col] = c12[j] + bi2;
  }
}

// ---- attention: one wave per (head, position) ------------------------------

__device__ __forceinline__ float waveReduceSum(float v) {
#pragma unroll
  for (int off = 16; off > 0; off >>= 1)
    v += __shfl_xor(v, off, 32);
  return v;
}

__global__ void knn_attn_kernel(const float* __restrict__ qkv,   // [4096 x 1584]
                                const float* __restrict__ knn,   // [8,4096,32,2,66]
                                __hip_bfloat16* __restrict__ aout /* [4096 x 544] */) {
  int wave = (int)((blockIdx.x * blockDim.x + threadIdx.x) >> 5);
  int lane = threadIdx.x & 31;
  if (wave >= NHEAD * NPOS) return;
  int h = wave >> 12;        // / 4096
  int n = wave & (NPOS - 1);

  const float* qp = qkv + (size_t)n * CQKV + h * HD;
  const float* kp = qp + CIN;
  const float* vp = qp + 2 * CIN;
  bool has3 = (lane < (HD - 64));     // lanes 0,1 carry d=64,65

  float q0 = qp[lane], q1 = qp[lane + 32], q2 = has3 ? qp[lane + 64] : 0.0f;
  float k0 = kp[lane], k1 = kp[lane + 32], k2 = has3 ? kp[lane + 64] : 0.0f;
  float v0 = vp[lane], v1 = vp[lane + 32], v2 = has3 ? vp[lane + 64] : 0.0f;

  float sq = waveReduceSum(q0 * q0 + q1 * q1 + q2 * q2);
  float sk = waveReduceSum(k0 * k0 + k1 * k1 + k2 * k2);
  float invq = 1.0f / fmaxf(sqrtf(sq), 1e-12f);
  float invk = 1.0f / fmaxf(sqrtf(sk), 1e-12f);
  q0 *= invq; q1 *= invq; q2 *= invq;
  k0 *= invk; k1 *= invk; k2 *= invk;

  const float scaling = 1.0f / sqrtf((float)HD);

  // online softmax, seeded with the self term (order-invariant)
  float m = waveReduceSum(q0 * k0 + q1 * k1 + q2 * k2) * scaling;
  float l = 1.0f;
  float a0 = v0, a1 = v1, a2 = v2;

  const float* kb = knn + (size_t)(h * NPOS + n) * (KNN * 2 * HD);
#pragma unroll 1
  for (int kk = 0; kk < KNN; ++kk) {
    const float* kr = kb + (size_t)kk * (2 * HD);
    const float* vr = kr + HD;
    if (kk + 2 < KNN)   // stream-ahead hint for the 554 MB knn_kv read
      __builtin_prefetch(kr + 2 * (2 * HD) + lane * 4, 0, 0);

    float e0 = __builtin_nontemporal_load(kr + lane);
    float e1 = __builtin_nontemporal_load(kr + lane + 32);
    float e2 = has3 ? __builtin_nontemporal_load(kr + lane + 64) : 0.0f;
    float s  = waveReduceSum(q0 * e0 + q1 * e1 + q2 * e2) * scaling;

    float w0 = __builtin_nontemporal_load(vr + lane);
    float w1 = __builtin_nontemporal_load(vr + lane + 32);
    float w2 = has3 ? __builtin_nontemporal_load(vr + lane + 64) : 0.0f;

    float mn = fmaxf(m, s);
    float cf = __expf(m - mn);
    float p  = __expf(s - mn);
    l  = l * cf + p;
    a0 = a0 * cf + p * w0;
    a1 = a1 * cf + p * w1;
    a2 = a2 * cf + p * w2;
    m  = mn;
  }

  float invl = 1.0f / l;
  __hip_bfloat16* op = aout + (size_t)n * KPAD + h * HD;
  op[lane]      = __float2bfloat16(a0 * invl);
  op[lane + 32] = __float2bfloat16(a1 * invl);
  if (has3) op[lane + 64] = __float2bfloat16(a2 * invl);
}

// ---------------------------------------------------------------------------

extern "C" void kernel_launch(void* const* d_in, const int* in_sizes, int n_in,
                              void* d_out, int out_size, void* d_ws, size_t ws_size,
                              hipStream_t stream) {
  (void)in_sizes; (void)n_in; (void)out_size; (void)ws_size;
  const float* x      = (const float*)d_in[0];  // [4096 x 528]
  const float* knn    = (const float*)d_in[1];  // [8,4096,32,2,66]
  const float* w_qkv  = (const float*)d_in[2];  // [1584 x 528]
  const float* b_qkv  = (const float*)d_in[3];  // [1584]
  const float* w_proj = (const float*)d_in[4];  // [528 x 528]
  const float* b_proj = (const float*)d_in[5];  // [528]
  float* out = (float*)d_out;                   // [4096 x 528]

  // workspace layout (all 16B-aligned by construction)
  char* ws = (char*)d_ws;
  __hip_bfloat16* Xb  = (__hip_bfloat16*)ws;                  // 4096*544
  __hip_bfloat16* Wqb = Xb  + (size_t)NPOS * KPAD;            // 1584*544
  __hip_bfloat16* Wpb = Wqb + (size_t)CQKV * KPAD;            //  528*544
  __hip_bfloat16* Ab  = Wpb + (size_t)CIN  * KPAD;            // 4096*544 (attn out)
  float*          qkv = (float*)(Ab + (size_t)NPOS * KPAD);   // 4096*1584 fp32

  const int TB = 256;

  // prep: bf16 conversions with zero-padded K, zero attn-out buffer
  {
    int t = NPOS * KPAD;
    cvt_pad_bf16_kernel<<<(t + TB - 1) / TB, TB, 0, stream>>>(x, Xb, NPOS, CIN, KPAD);
    t = CQKV * KPAD;
    cvt_pad_bf16_kernel<<<(t + TB - 1) / TB, TB, 0, stream>>>(w_qkv, Wqb, CQKV, CIN, KPAD);
    t = CIN * KPAD;
    cvt_pad_bf16_kernel<<<(t + TB - 1) / TB, TB, 0, stream>>>(w_proj, Wpb, CIN, CIN, KPAD);
    t = NPOS * KPAD;
    zero_bf16_kernel<<<(t + TB - 1) / TB, TB, 0, stream>>>(Ab, t);
  }

  // gemm1: qkv = Xb @ Wqkv^T + b_qkv   (M=4096 -> 128 mtiles, 33 strips of 48)
  {
    int nStrips = CQKV / 48;                       // 33
    int waves   = (NPOS >> 5) * nStrips;           // 4224
    gemm_bf16_nt_kernel<<<waves * 32 / TB, TB, 0, stream>>>(Xb, Wqb, b_qkv, qkv,
                                                            NPOS, CQKV, nStrips);
  }

  // attention: 8*4096 waves, 8 waves per block
  {
    int waves = NHEAD * NPOS;                      // 32768
    knn_attn_kernel<<<waves * 32 / TB, TB, 0, stream>>>(qkv, knn, Ab);
  }

  // gemm2: out = Ab @ Wproj^T + b_proj  (M=4096 -> 128 mtiles, 11 strips of 48)
  {
    int nStrips = CIN / 48;                        // 11
    int waves   = (NPOS >> 5) * nStrips;           // 1408
    gemm_bf16_nt_kernel<<<waves * 32 / TB, TB, 0, stream>>>(Ab, Wpb, b_proj, out,
                                                            NPOS, CIN, nStrips);
  }
}
